// GatedRecurrenceBlock_57019985822491
// MI455X (gfx1250) — compile-verified, matrix operand order checked by reading, not run on hardware
//
#include <hip/hip_runtime.h>
#include <hip/hip_bf16.h>

typedef __attribute__((ext_vector_type(16))) __bf16 v16bf;
typedef __attribute__((ext_vector_type(8)))  __bf16 v8bf;
typedef __attribute__((ext_vector_type(8)))  float  v8f;

constexpr int H  = 1024;
constexpr int B  = 8;
constexpr int S  = 2048;
constexpr int R  = B * S;          // 16384 GEMM rows
constexpr int NWG = 64;            // scan workgroups (persistent)
constexpr int NSL = H / NWG;       // 16 output columns per scan WG

__device__ inline v8f wmma_bf16(v16bf a, v16bf b, v8f c) {
  return __builtin_amdgcn_wmma_f32_16x16x32_bf16(false, a, false, b, (short)0, c,
                                                 false, false);
}

__device__ inline v16bf cat16(v8bf lo, v8bf hi) {
  return __builtin_shufflevector(lo, hi, 0, 1, 2, 3, 4, 5, 6, 7,
                                 8, 9, 10, 11, 12, 13, 14, 15);
}

// Fragment-order swizzle for 16-bit B operands (K x N, 16x32-wide wmma tiles).
// Element (k, n) lives at block(ntile,ktile), lane = n%16 | ((k>>3)&1)<<4,
// e = (k%32>=16 ? 8:0) + k%8.  Each lane's 16 elements are contiguous (32 B).
__device__ __host__ inline size_t swz(int K, int n, int k) {
  int kl   = k & 31;
  int lane = (n & 15) | (((kl >> 3) & 1) << 4);
  int e    = ((kl & 16) ? 8 : 0) | (kl & 7);
  size_t blk = (size_t)(n >> 4) * (K >> 5) + (k >> 5);
  return blk * 512 + (size_t)lane * 16 + e;
}

__device__ inline float sigmoidf_(float x) { return 1.f / (1.f + __expf(-x)); }
__device__ inline float siluf_(float x)    { return x * sigmoidf_(x); }

// ---------------------------------------------------------------- weights->bf16 (swizzled)
__global__ void k_convert(const float* __restrict__ W_in,
                          const float* __restrict__ W_rec,
                          const float* __restrict__ W_gate,
                          const float* __restrict__ W_out,
                          __bf16* __restrict__ Winz,
                          __bf16* __restrict__ Wcz,
                          __bf16* __restrict__ Woutz) {
  size_t i      = (size_t)blockIdx.x * blockDim.x + threadIdx.x;
  size_t stride = (size_t)gridDim.x * blockDim.x;
  const int N3 = 3 * H;
  for (size_t j = i; j < (size_t)H * N3; j += stride) {
    int k = (int)(j / N3), n = (int)(j % N3);
    Winz[swz(H, n, k)] = (__bf16)W_in[j];
  }
  for (size_t j = i; j < (size_t)H * N3; j += stride) {
    int k = (int)(j / N3), n = (int)(j % N3);
    float v = (n < 2 * H) ? W_gate[(size_t)k * 2 * H + n]
                          : W_rec[(size_t)k * H + (n - 2 * H)];
    Wcz[swz(H, n, k)] = (__bf16)v;
  }
  for (size_t j = i; j < (size_t)H * H; j += stride) {
    int k = (int)(j / H), n = (int)(j % H);
    Woutz[swz(H, n, k)] = (__bf16)W_out[j];
  }
}

// ---------------------------------------------------------------- scan state init
__global__ void k_init(const float* __restrict__ h0,
                       __bf16* __restrict__ hbf,     // [16][H], rows 8..15 = 0
                       float* __restrict__ stats,    // 2 x 8 x {sum,sumsq}
                       unsigned* __restrict__ bar) {
  int i = blockIdx.x * blockDim.x + threadIdx.x;
  int stride = gridDim.x * blockDim.x;
  for (int j = i; j < 16 * H; j += stride) {
    int r = j / H, c = j % H;
    hbf[j] = (__bf16)((r < B) ? h0[r * H + c] : 0.f);
  }
  if (i < 32) stats[i] = 0.f;
  if (i < 2)  bar[i]  = 0u;
}

// ---------------------------------------------------------------- input GEMM fused
// P[s,b,n] = silu(u)*silu(v) + w  from feats = x @ W_in + b_in
__global__ __launch_bounds__(256) void k_gemm_in(const float*  __restrict__ x,
                                                 const __bf16* __restrict__ Winz,
                                                 const float*  __restrict__ b_in,
                                                 float*        __restrict__ P) {
  __shared__ alignas(32) __bf16 As[64][32];   // 4 KB
  const int tid  = threadIdx.x;
  const int lane = tid & 31;
  const int wave = tid >> 5;           // 8 waves
  const int wm   = wave & 3;           // 4 M-tiles of 16 rows
  const int wn   = wave >> 2;          // 2 N-halves of 32 cols
  const int tile_r = blockIdx.x * 64;
  const int n0     = blockIdx.y * 64;

  v8f acc[3][2];
  for (int sl = 0; sl < 3; ++sl)
    for (int nt = 0; nt < 2; ++nt)
      for (int e = 0; e < 8; ++e) acc[sl][nt][e] = 0.f;

  const int sr = tid >> 2;             // A staging: row per thread
  const int sc = (tid & 3) * 8;        // 8-col segment

  for (int k0 = 0; k0 < H; k0 += 32) {
    __syncthreads();
    {
      const float4* xp = (const float4*)&x[(size_t)(tile_r + sr) * H + k0 + sc];
      float4 f0 = xp[0], f1 = xp[1];
      v8bf o;
      o[0] = (__bf16)f0.x; o[1] = (__bf16)f0.y; o[2] = (__bf16)f0.z; o[3] = (__bf16)f0.w;
      o[4] = (__bf16)f1.x; o[5] = (__bf16)f1.y; o[6] = (__bf16)f1.z; o[7] = (__bf16)f1.w;
      *(v8bf*)&As[sr][sc] = o;
    }
    __syncthreads();

    const __bf16* ap = &As[wm * 16 + (lane & 15)][(lane >> 4) << 3];
    v16bf a = cat16(*(const v8bf*)ap, *(const v8bf*)(ap + 16));

    for (int nt = 0; nt < 2; ++nt) {
      for (int sl = 0; sl < 3; ++sl) {
        int ng = sl * H + n0 + wn * 32 + nt * 16;
        const __bf16* bp =
            Winz + ((size_t)(ng >> 4) * (H >> 5) + (k0 >> 5)) * 512 + lane * 16;
        v16bf b = *(const v16bf*)bp;
        acc[sl][nt] = wmma_bf16(a, b, acc[sl][nt]);
      }
    }
  }

  for (int nt = 0; nt < 2; ++nt) {
    int n = n0 + wn * 32 + nt * 16 + (lane & 15);
    float bu = b_in[n], bv = b_in[H + n], bw = b_in[2 * H + n];
    for (int e = 0; e < 8; ++e) {
      int m  = e + (lane >> 4) * 8;
      int r  = tile_r + wm * 16 + m;
      int bb = r / S, ss = r % S;     // x rows are [B,S] flattened
      float u = acc[0][nt][e] + bu;
      float v = acc[1][nt][e] + bv;
      float w = acc[2][nt][e] + bw;
      P[((size_t)ss * B + bb) * H + n] = siluf_(u) * siluf_(v) + w;
    }
  }
}

// ---------------------------------------------------------------- grid barrier
__device__ inline void grid_barrier(unsigned* bar) {
  __syncthreads();
  __threadfence();
  if (threadIdx.x == 0) {
    unsigned gen = __atomic_load_n(&bar[1], __ATOMIC_RELAXED);
    if (atomicAdd(&bar[0], 1u) == NWG - 1) {
      bar[0] = 0;
      __threadfence();
      atomicAdd(&bar[1], 1u);
    } else {
      while (__atomic_load_n(&bar[1], __ATOMIC_RELAXED) == gen)
        __builtin_amdgcn_s_sleep(1);
    }
  }
  __syncthreads();
}

// ---------------------------------------------------------------- persistent scan
__global__ __launch_bounds__(128) void k_scan(const __bf16* __restrict__ Wcz,
                                              const float*  __restrict__ b_gate,
                                              const float*  __restrict__ b_rec,
                                              const float*  __restrict__ P,
                                              const float*  __restrict__ ln_g,
                                              const float*  __restrict__ ln_b,
                                              __bf16* __restrict__ hbf,     // [16][H]
                                              __bf16* __restrict__ Hs,      // [S,B,H]
                                              float*  __restrict__ stats,   // [2][8][2]
                                              unsigned* __restrict__ bar,
                                              float* __restrict__ final_h) {
  __shared__ float red[4][3][32][8];   // 12 KB cross-wave partials
  const int tid  = threadIdx.x;
  const int lane = tid & 31;
  const int wave = tid >> 5;           // 4 waves split K
  const int n0   = blockIdx.x * NSL;   // 16 columns owned by this WG

  for (int t = 0; t < S; ++t) {
    // prefetch next timestep's candidate slice into L2
    if (t + 1 < S && tid < B)
      __builtin_prefetch(&P[((size_t)(t + 1) * B + tid) * H + n0], 0, 1);

    v8f acc[3];
    for (int sl = 0; sl < 3; ++sl)
      for (int e = 0; e < 8; ++e) acc[sl][e] = 0.f;

    const int kbase = wave * (H / 4);
    const int arow  = lane & 15;
    for (int kc = 0; kc < H / 4; kc += 32) {
      int k0 = kbase + kc;
      const __bf16* ap = hbf + (size_t)arow * H + k0 + ((lane >> 4) << 3);
      v16bf a = cat16(*(const v8bf*)ap, *(const v8bf*)(ap + 16));
      for (int sl = 0; sl < 3; ++sl) {
        const __bf16* bp =
            Wcz + ((size_t)((sl * H + n0) >> 4) * (H >> 5) + (k0 >> 5)) * 512 +
            lane * 16;
        v16bf b = *(const v16bf*)bp;
        acc[sl] = wmma_bf16(a, b, acc[sl]);
      }
    }

    for (int sl = 0; sl < 3; ++sl)
      for (int e = 0; e < 8; ++e) red[wave][sl][lane][e] = acc[sl][e];
    __syncthreads();

    float nh[8];
    if (wave == 0 && lane < 16) {       // rows m=0..7 live in lanes 0..15, e=0..7
      int n = n0 + lane;
      float lsum[8], lsq[8];
      for (int e = 0; e < 8; ++e) {
        float y0 = red[0][0][lane][e] + red[1][0][lane][e] + red[2][0][lane][e] + red[3][0][lane][e];
        float y1 = red[0][1][lane][e] + red[1][1][lane][e] + red[2][1][lane][e] + red[3][1][lane][e];
        float y2 = red[0][2][lane][e] + red[1][2][lane][e] + red[2][2][lane][e] + red[3][2][lane][e];
        float gv = y0 + b_gate[n];
        float gt = y1 + b_gate[H + n];
        float rc = y2 + b_rec[n];
        float gate = sigmoidf_(gv) * siluf_(gt);
        float p = P[((size_t)t * B + e) * H + n];
        float val = gate * rc + (1.f - gate) * p;
        nh[e] = val;
        lsum[e] = val;
        lsq[e]  = val * val;
      }
      for (int off = 8; off >= 1; off >>= 1)
        for (int e = 0; e < 8; ++e) {
          lsum[e] += __shfl_xor(lsum[e], off, 32);
          lsq[e]  += __shfl_xor(lsq[e],  off, 32);
        }
      if (lane == 0) {
        float* sb = stats + (t & 1) * 16;
        for (int e = 0; e < 8; ++e) {
          atomicAdd(&sb[e * 2 + 0], lsum[e]);
          atomicAdd(&sb[e * 2 + 1], lsq[e]);
        }
      }
    }

    grid_barrier(bar);                  // stats complete

    if (wave == 0 && lane < 16) {
      const float* sb = stats + (t & 1) * 16;
      int n = n0 + lane;
      float g_ = ln_g[n], bt_ = ln_b[n];
      for (int e = 0; e < 8; ++e) {
        float mean = sb[e * 2 + 0] * (1.f / H);
        float var  = sb[e * 2 + 1] * (1.f / H) - mean * mean;
        float rstd = rsqrtf(var + 1e-5f);
        float y = (nh[e] - mean) * rstd * g_ + bt_;
        hbf[(size_t)e * H + n]              = (__bf16)y;
        Hs[((size_t)t * B + e) * H + n]     = (__bf16)y;
        if (t == S - 1) final_h[(size_t)e * H + n] = y;
      }
    }
    // reset the other stats buffer for step t+1 (its readers finished at t-1)
    if (blockIdx.x == 0 && tid == 0) {
      float* ob = stats + ((t + 1) & 1) * 16;
      for (int i = 0; i < 16; ++i) ob[i] = 0.f;
    }

    grid_barrier(bar);                  // hbf / stats-reset visible
  }
}

// ---------------------------------------------------------------- output GEMM
__global__ __launch_bounds__(256) void k_gemm_out(const __bf16* __restrict__ Hs,
                                                  const __bf16* __restrict__ Woutz,
                                                  const float*  __restrict__ b_out,
                                                  float*        __restrict__ out) {
  __shared__ alignas(32) __bf16 As[64][32];
  const int tid  = threadIdx.x;
  const int lane = tid & 31;
  const int wave = tid >> 5;
  const int wm   = wave & 3;
  const int wn   = wave >> 2;
  const int tile_r = blockIdx.x * 64;
  const int n0     = blockIdx.y * 64;

  v8f acc[2];
  for (int nt = 0; nt < 2; ++nt)
    for (int e = 0; e < 8; ++e) acc[nt][e] = 0.f;

  const int sr = tid >> 2;
  const int sc = (tid & 3) * 8;

  for (int k0 = 0; k0 < H; k0 += 32) {
    __syncthreads();
    *(v8bf*)&As[sr][sc] =
        *(const v8bf*)&Hs[(size_t)(tile_r + sr) * H + k0 + sc];
    __syncthreads();

    const __bf16* ap = &As[wm * 16 + (lane & 15)][(lane >> 4) << 3];
    v16bf a = cat16(*(const v8bf*)ap, *(const v8bf*)(ap + 16));

    for (int nt = 0; nt < 2; ++nt) {
      int ng = n0 + wn * 32 + nt * 16;
      const __bf16* bp =
          Woutz + ((size_t)(ng >> 4) * (H >> 5) + (k0 >> 5)) * 512 + lane * 16;
      v16bf b = *(const v16bf*)bp;
      acc[nt] = wmma_bf16(a, b, acc[nt]);
    }
  }

  for (int nt = 0; nt < 2; ++nt) {
    int n = n0 + wn * 32 + nt * 16 + (lane & 15);
    float bo = b_out[n];
    for (int e = 0; e < 8; ++e) {
      int m  = e + (lane >> 4) * 8;
      int r  = tile_r + wm * 16 + m;     // Hs rows are [S,B] flattened
      int ss = r >> 3, bb = r & 7;
      out[((size_t)bb * S + ss) * H + n] = acc[nt][e] + bo;
    }
  }
}

// ---------------------------------------------------------------- launch
extern "C" void kernel_launch(void* const* d_in, const int* in_sizes, int n_in,
                              void* d_out, int out_size, void* d_ws, size_t ws_size,
                              hipStream_t stream) {
  const float* x      = (const float*)d_in[0];
  const float* h0     = (const float*)d_in[1];
  const float* W_in   = (const float*)d_in[2];
  const float* b_in   = (const float*)d_in[3];
  const float* W_rec  = (const float*)d_in[4];
  const float* b_rec  = (const float*)d_in[5];
  const float* W_gate = (const float*)d_in[6];
  const float* b_gate = (const float*)d_in[7];
  const float* W_out  = (const float*)d_in[8];
  const float* b_out  = (const float*)d_in[9];
  const float* ln_g   = (const float*)d_in[10];
  const float* ln_b   = (const float*)d_in[11];
  float* out = (float*)d_out;
  float* final_h = out + (size_t)B * S * H;

  char* ws = (char*)d_ws;
  size_t cur = 0;
  auto carve = [&](size_t bytes) {
    void* p = ws + cur;
    cur = (cur + bytes + 255) & ~(size_t)255;
    return p;
  };
  __bf16*   Winz  = (__bf16*)carve((size_t)H * 3 * H * sizeof(__bf16));
  __bf16*   Wcz   = (__bf16*)carve((size_t)H * 3 * H * sizeof(__bf16));
  __bf16*   Woutz = (__bf16*)carve((size_t)H * H * sizeof(__bf16));
  float*    P     = (float*)carve((size_t)S * B * H * sizeof(float));
  __bf16*   Hs    = (__bf16*)carve((size_t)S * B * H * sizeof(__bf16));
  __bf16*   hbf   = (__bf16*)carve((size_t)16 * H * sizeof(__bf16));
  float*    stats = (float*)carve(32 * sizeof(float));
  unsigned* bar   = (unsigned*)carve(2 * sizeof(unsigned));

  k_convert<<<1024, 256, 0, stream>>>(W_in, W_rec, W_gate, W_out, Winz, Wcz, Woutz);
  k_init<<<64, 256, 0, stream>>>(h0, hbf, stats, bar);

  dim3 gIn(R / 64, H / 64);
  k_gemm_in<<<gIn, 256, 0, stream>>>(x, Winz, b_in, P);

  k_scan<<<NWG, 128, 0, stream>>>(Wcz, b_gate, b_rec, P, ln_g, ln_b,
                                  hbf, Hs, stats, bar, final_h);

  dim3 gOut(R / 64, H / 64);
  k_gemm_out<<<gOut, 256, 0, stream>>>(Hs, Woutz, b_out, out);
}
